// MaskingModel_66030827209087
// MI455X (gfx1250) — matrix operand order repaired and checked by reading, not compiled, not run on hardware
//
#include <hip/hip_runtime.h>
#include <hip/hip_bf16.h>
#include <math.h>

// Problem constants (match reference)
#define BGR   64          // graphs
#define NPG   4096        // nodes per graph
#define MINS  8           // instructions
#define DN    512         // node feature dim (K of big GEMM)
#define DQ    256         // question dim (N of big GEMM)
#define NTOT  (BGR*NPG)   // 262144 nodes
#define KSEL  1229        // ceil(0.3*4096)

typedef __attribute__((ext_vector_type(16))) __bf16 v16bf;
typedef __attribute__((ext_vector_type(8)))  float  v8f;

// ---------------------------------------------------------------------------
// Kernel 0: pack Wn (f32 [512][256]) into bf16x2 (hi/lo) WMMA-B fragments.
// Fragment (ktile 0..15, ntile 0..15) = 32 lanes x 16 bf16 contiguous.
// Lane L holds col n = ntile*16 + (L&15), K = ktile*32 + 16*(L>>4) + j, j=0..15
// ---------------------------------------------------------------------------
__global__ void pack_wn_kernel(const float* __restrict__ Wn,
                               __bf16* __restrict__ wn_hi,
                               __bf16* __restrict__ wn_lo) {
    const int g    = blockIdx.x * blockDim.x + threadIdx.x;   // 0..8191
    const int frag = g >> 5;
    const int lane = g & 31;
    const int ktile = frag >> 4;
    const int ntile = frag & 15;
    const int col   = ntile * 16 + (lane & 15);
    const int kbase = ktile * 32 + (lane >> 4) * 16;
    __bf16* dh = wn_hi + (size_t)g * 16;
    __bf16* dl = wn_lo + (size_t)g * 16;
#pragma unroll
    for (int j = 0; j < 16; ++j) {
        const float wv = Wn[(size_t)(kbase + j) * DQ + col];
        const __bf16 hi = (__bf16)wv;
        dh[j] = hi;
        dl[j] = (__bf16)(wv - (float)hi);
    }
}

__device__ __forceinline__ float gelu_exact(float v) {
    return 0.5f * v * (1.0f + erff(v * 0.7071067811865476f));
}

// ---------------------------------------------------------------------------
// Kernel 1: fused  h = gelu(x @ Wn + bn)  and  scores[b,m,n] = <u[m,b,:], h>
// Block = 256 threads (8 waves), owns 32 nodes. bf16x3 split WMMA, f32 acc:
//   x*w ~= xh*wh + xl*wh + xh*wl   (drops only ~2^-16 xl*wl term)
// ---------------------------------------------------------------------------
__global__ void __launch_bounds__(256)
gemm_scores_kernel(const float* __restrict__ x,
                   const float* __restrict__ u,
                   const v16bf* __restrict__ wnp_hi,  // packed Wn hi fragments
                   const v16bf* __restrict__ wnp_lo,  // packed Wn lo fragments
                   const float* __restrict__ bn,
                   float* __restrict__ scores) {      // [B][M][NPG]
    __shared__ float xs[32][36];       // x tile (f32, padded)
    __shared__ float hs[32][260];      // gelu output tile (padded)
    __shared__ float us[8][260];       // u rows for this graph (padded)

    const int tid   = threadIdx.x;
    const int mbase = blockIdx.x * 32;             // global node base
    const int b     = blockIdx.x >> 7;             // 128 blocks per graph
    const int nloc  = (blockIdx.x & 127) * 32;     // node-in-graph base

    // stage u[m, b, :] into LDS (8 x 256 f32)
#pragma unroll
    for (int j = 0; j < 8; ++j) {
        const int e = tid * 8 + j;                 // 0..2047
        const int m = e >> 8, d = e & 255;
        us[m][d] = u[((size_t)m * BGR + b) * DQ + d];
    }

    const int w    = tid >> 5;
    const int lane = tid & 31;
    const int mt   = w >> 2;                       // 0..1 : 16-row tile
    const int nq   = w & 3;                        // 0..3 : 4 col-tiles each
    const int arow = mt * 16 + (lane & 15);        // A-fragment row in tile
    const int ac0  = (lane >> 4) * 16;             // A-fragment K sub-offset

    v8f acc[4] = {};

    const int xrow = tid >> 3;                     // cooperative x-tile load
    const int xc4  = (tid & 7) * 4;

    for (int kk = 0; kk < 16; ++kk) {              // K chunks of 32
        const float4 xv = *(const float4*)(x + (size_t)(mbase + xrow) * DN
                                             + kk * 32 + xc4);
        *(float4*)&xs[xrow][xc4] = xv;
        __syncthreads();

        // Build bf16 hi/lo A fragments, contiguous K = 16*(lane>>4) + j (matches packed B)
        v16bf ah, al;
#pragma unroll
        for (int j = 0; j < 8; ++j) {
            const float f0 = xs[arow][ac0 + j];
            const float f1 = xs[arow][ac0 + 8 + j];
            const __bf16 h0 = (__bf16)f0;
            const __bf16 h1 = (__bf16)f1;
            ah[j]     = h0;
            ah[j + 8] = h1;
            al[j]     = (__bf16)(f0 - (float)h0);
            al[j + 8] = (__bf16)(f1 - (float)h1);
        }
#pragma unroll
        for (int j = 0; j < 4; ++j) {
            const int nt = nq * 4 + j;
            const size_t fidx = (size_t)(kk * 16 + nt) * 32 + lane;
            const v16bf bh = wnp_hi[fidx];
            const v16bf bl = wnp_lo[fidx];
            acc[j] = __builtin_amdgcn_wmma_f32_16x16x32_bf16(
                         false, ah, false, bh, (short)0, acc[j], false, false);
            acc[j] = __builtin_amdgcn_wmma_f32_16x16x32_bf16(
                         false, al, false, bh, (short)0, acc[j], false, false);
            acc[j] = __builtin_amdgcn_wmma_f32_16x16x32_bf16(
                         false, ah, false, bl, (short)0, acc[j], false, false);
        }
        __syncthreads();
    }

    // Epilogue: bias + exact GELU -> LDS h tile
#pragma unroll
    for (int j = 0; j < 4; ++j) {
        const int col = (nq * 4 + j) * 16 + (lane & 15);
        const float bias = bn[col];
#pragma unroll
        for (int r = 0; r < 8; ++r) {
            const int rr = mt * 16 + r + (lane >> 4) * 8;   // C layout row
            hs[rr][col] = gelu_exact(acc[j][r] + bias);
        }
    }
    __syncthreads();

    // Tiny score GEMM: one (node, m) pair per thread, f32 VALU
    const int node = tid >> 3;
    const int m    = tid & 7;
    float s = 0.f;
#pragma unroll 8
    for (int d = 0; d < DQ; ++d) s += hs[node][d] * us[m][d];
    scores[((size_t)(b * MINS + m)) * NPG + nloc + node] = s;
}

// ---------------------------------------------------------------------------
// Kernel 2: per-(b,m) softmax over n, summed over m -> gate[b][n]
// ---------------------------------------------------------------------------
__global__ void __launch_bounds__(256)
softmax_gate_kernel(const float* __restrict__ sc, float* __restrict__ gate) {
    const int b = blockIdx.x, tid = threadIdx.x;
    __shared__ float red[256];
    __shared__ float mxs[8], rden[8];

    for (int m = 0; m < MINS; ++m) {
        const float* p = sc + ((size_t)(b * MINS + m)) * NPG;
        float lm = -3.4e38f;
#pragma unroll
        for (int i = 0; i < 16; ++i) lm = fmaxf(lm, p[tid + i * 256]);
        red[tid] = lm; __syncthreads();
        for (int s = 128; s > 0; s >>= 1) {
            if (tid < s) red[tid] = fmaxf(red[tid], red[tid + s]);
            __syncthreads();
        }
        const float mxm = red[0]; __syncthreads();

        float ls = 0.f;
#pragma unroll
        for (int i = 0; i < 16; ++i) ls += expf(p[tid + i * 256] - mxm);
        red[tid] = ls; __syncthreads();
        for (int s = 128; s > 0; s >>= 1) {
            if (tid < s) red[tid] += red[tid + s];
            __syncthreads();
        }
        if (tid == 0) { mxs[m] = mxm; rden[m] = 1.f / red[0]; }
        __syncthreads();
    }

#pragma unroll 1
    for (int i = 0; i < 16; ++i) {
        const int n = tid + i * 256;
        float g = 0.f;
#pragma unroll
        for (int m = 0; m < MINS; ++m)
            g += expf(sc[((size_t)(b * MINS + m)) * NPG + n] - mxs[m]) * rden[m];
        gate[(size_t)b * NPG + n] = g;
    }
}

// ---------------------------------------------------------------------------
// Kernel 3: exact K-th largest per graph via 4-pass radix select on
// order-preserving uint keys; mask = (gate >= kth) (ties included).
// ---------------------------------------------------------------------------
__global__ void __launch_bounds__(256)
topk_mask_kernel(const float* __restrict__ gate, float* __restrict__ out) {
    const int b = blockIdx.x, tid = threadIdx.x;
    __shared__ unsigned keys[NPG];
    __shared__ unsigned hist[256];
    __shared__ unsigned sh_prefix, sh_k;

#pragma unroll
    for (int i = 0; i < 16; ++i) {
        const int n = tid + i * 256;
        const unsigned bits = __float_as_uint(gate[(size_t)b * NPG + n]);
        keys[n] = (bits & 0x80000000u) ? ~bits : (bits | 0x80000000u);
    }
    if (tid == 0) { sh_prefix = 0u; sh_k = (unsigned)KSEL; }
    __syncthreads();

    for (int pass = 0; pass < 4; ++pass) {
        const int shift = 24 - 8 * pass;
        hist[tid] = 0u;
        __syncthreads();
        const unsigned maskknown = (pass == 0) ? 0u : (0xFFFFFFFFu << (shift + 8));
        const unsigned pfx = sh_prefix;
#pragma unroll
        for (int i = 0; i < 16; ++i) {
            const unsigned uk = keys[tid + i * 256];
            if ((uk & maskknown) == pfx)
                atomicAdd(&hist[(uk >> shift) & 255u], 1u);
        }
        __syncthreads();
        if (tid == 0) {
            unsigned k = sh_k, cum = 0u;
            for (int bin = 255; bin >= 0; --bin) {
                const unsigned c = hist[bin];
                if (cum + c >= k) {
                    sh_prefix = pfx | ((unsigned)bin << shift);
                    sh_k = k - cum;
                    break;
                }
                cum += c;
            }
        }
        __syncthreads();
    }

    const unsigned kth = sh_prefix;   // exact key of K-th largest gate
#pragma unroll
    for (int i = 0; i < 16; ++i) {
        const int n = tid + i * 256;
        out[(size_t)b * NPG + n] = (keys[n] >= kth) ? 1.0f : 0.0f;
    }
}

// ---------------------------------------------------------------------------
extern "C" void kernel_launch(void* const* d_in, const int* in_sizes, int n_in,
                              void* d_out, int out_size, void* d_ws, size_t ws_size,
                              hipStream_t stream) {
    const float* x  = (const float*)d_in[0];   // [N, DN]
    const float* u  = (const float*)d_in[1];   // [M, B, DQ]
    const float* Wn = (const float*)d_in[2];   // [DN, DQ]
    const float* bn = (const float*)d_in[3];   // [DQ]
    // d_in[4] batch, d_in[5] edge_index: unused by forward

    char* ws = (char*)d_ws;
    const size_t wn_bytes = (size_t)DN * DQ * 2;                     // 256 KB
    __bf16* wn_hi  = (__bf16*)ws;
    __bf16* wn_lo  = (__bf16*)(ws + wn_bytes);
    float*  scores = (float*)(ws + 2 * wn_bytes);                    // 8 MB
    float*  gate   = (float*)(ws + 2 * wn_bytes
                                 + (size_t)BGR * MINS * NPG * 4);    // 1 MB
    float*  out    = (float*)d_out;

    pack_wn_kernel<<<32, 256, 0, stream>>>(Wn, wn_hi, wn_lo);
    gemm_scores_kernel<<<NTOT / 32, 256, 0, stream>>>(
        x, u, (const v16bf*)wn_hi, (const v16bf*)wn_lo, bn, scores);
    softmax_gate_kernel<<<BGR, 256, 0, stream>>>(scores, gate);
    topk_mask_kernel<<<BGR, 256, 0, stream>>>(gate, out);
}
